// SpatialAttentionVer1_36215164240421
// MI455X (gfx1250) — matrix operand order repaired and checked by reading, not compiled, not run on hardware
//
#include <hip/hip_runtime.h>

typedef __attribute__((ext_vector_type(16))) __bf16 v16bf;
typedef __attribute__((ext_vector_type(8)))  float  v8f;
typedef __attribute__((ext_vector_type(4)))  float  v4f;
typedef __attribute__((ext_vector_type(2)))  float  v2f;
typedef __attribute__((ext_vector_type(4)))  unsigned int v4u;

#define BSZ   128
#define CDIM  272
#define TDIM  1024
#define DDIM  270
#define KL    1024        // 32*32 frequency pairs
#define CP    288         // padded C (multiple of 32)
#define DP    288         // padded D (multiple of 16, 18 tiles)
#define XPITCH 296        // LDS row pitch in bf16 units (592 B, 16-B aligned)

__device__ __forceinline__ unsigned short f2bf(float f) {
  unsigned int u = __float_as_uint(f);
  u += 0x7FFFu + ((u >> 16) & 1u);
  return (unsigned short)(u >> 16);
}

union BF16x16 { v16bf v; v4u q[2]; };

// ---------------------------------------------------------------------------
// Stage 1: cs/sn[kl][c] = cos/sin(2*pi*(k*x_c + l*y_c)), zero-padded c>=272
// ---------------------------------------------------------------------------
__global__ void sa_phase_kernel(const float* __restrict__ loc,
                                float* __restrict__ cs, float* __restrict__ sn) {
  int kl = blockIdx.x;          // 0..1023
  int c  = threadIdx.x;         // 0..287
  float co = 0.f, si = 0.f;
  if (c < CDIM) {
    float x = loc[2 * c], y = loc[2 * c + 1];
    float k = (float)(kl >> 5), l = (float)(kl & 31);
    float ph = 6.2831853071795864769f * (k * x + l * y);
    sincosf(ph, &si, &co);
  }
  cs[(size_t)kl * CP + c] = co;
  sn[(size_t)kl * CP + c] = si;
}

// ---------------------------------------------------------------------------
// Stage 2: A[d][c] = sum_kl z_re*cs + z_im*sn  via V_WMMA_F32_16X16X4_F32.
// One wave per 16x16 tile; 18x18 tiles over (DP, CP). Exact fp32 accumulate.
// Rows d>=270 are never read by softmax, so the z row index is simply clamped
// (padding rows of A hold duplicated-row garbage) -> no EXEC churn in the
// inner loop.
// A-frag (16x4 f32): lane l<16 row M=l, VGPR0=K(2h), VGPR1=K(2h+1), h=lane>>4.
// B-frag (4x16 f32): VGPR0=K(2h), VGPR1=K(2h+1), N=lane&15.
// ---------------------------------------------------------------------------
__global__ void sa_a_wmma_kernel(const float* __restrict__ zre,
                                 const float* __restrict__ zim,
                                 const float* __restrict__ cs,
                                 const float* __restrict__ sn,
                                 float* __restrict__ A) {
  int tile = blockIdx.x;              // 0..323
  int dt = tile / 18, ct = tile % 18;
  int d0 = dt * 16, c0 = ct * 16;
  int L = threadIdx.x, l = L & 15, h = L >> 4;
  int d = d0 + l;
  int dc = d < DDIM ? d : (DDIM - 1);   // clamp: padding rows ignored later
  const float* zr = zre + (size_t)dc * KL;
  const float* zi = zim + (size_t)dc * KL;
  int c = c0 + l;

  v8f acc = {};
  #pragma unroll 4
  for (int k0 = 0; k0 < KL; k0 += 4) {
    int ka = k0 + 2 * h;
    v2f za = *(const v2f*)(zr + ka);
    v2f zb = *(const v2f*)(zi + ka);
    v2f bc, bs;
    bc.x = cs[(size_t)ka * CP + c];
    bc.y = cs[(size_t)(ka + 1) * CP + c];
    bs.x = sn[(size_t)ka * CP + c];
    bs.y = sn[(size_t)(ka + 1) * CP + c];
    acc = __builtin_amdgcn_wmma_f32_16x16x4_f32(false, za, false, bc, (short)0, acc, false, false);
    acc = __builtin_amdgcn_wmma_f32_16x16x4_f32(false, zb, false, bs, (short)0, acc, false, false);
  }
  // C/D layout: VGPR r -> row d0 + r + 8h, col c0 + (lane&15)
  #pragma unroll
  for (int r = 0; r < 8; ++r)
    A[(size_t)(d0 + r + 8 * h) * CP + c] = acc[r];
}

// ---------------------------------------------------------------------------
// Stage 3: row softmax over c<272; emit bf16 weights, zero padding (c>=272,
// d>=270). One wave per d row.
// ---------------------------------------------------------------------------
__global__ void sa_softmax_kernel(const float* __restrict__ A,
                                  unsigned short* __restrict__ wBF) {
  int d = blockIdx.x;                 // 0..287
  int lane = threadIdx.x;             // 0..31
  unsigned short* wrow = wBF + (size_t)d * CP;
  if (d >= DDIM) {
    for (int c = lane; c < CP; c += 32) wrow[c] = 0;
    return;
  }
  const float* row = A + (size_t)d * CP;
  float m = -3.4e38f;
  for (int c = lane; c < CDIM; c += 32) m = fmaxf(m, row[c]);
  for (int off = 16; off; off >>= 1) m = fmaxf(m, __shfl_xor(m, off, 32));
  float e[9];
  float s = 0.f;
  int idx = 0;
  for (int c = lane; c < CDIM; c += 32) { float v = expf(row[c] - m); e[idx++] = v; s += v; }
  for (int off = 16; off; off >>= 1) s += __shfl_xor(s, off, 32);
  float inv = 1.f / s;
  idx = 0;
  for (int c = lane; c < CDIM; c += 32) wrow[c] = f2bf(e[idx++] * inv);
  for (int c = CDIM + lane; c < CP; c += 32) wrow[c] = 0;
}

// ---------------------------------------------------------------------------
// Stage 4: out[b,d,t] = sum_c w[d,c]*X[b,c,t] via v_wmma_f32_16x16x32_bf16.
// Block = (b, 64-t slab), 192 threads = 6 waves. X[c][t] fp32 -> LDS XT[t][c]
// bf16 (transposed so B-fragments are contiguous per lane). Each wave owns 3
// d-tiles x 4 t-tiles; K loop = 9 chunks of 32 over padded C=288.
// bf16 A 16x32 per lane: elems0-7 = K(8h..8h+7), elems8-15 = K(8h+16..8h+23).
// bf16 B 32x16 per lane: elems0-7 = K(8h..8h+7), elems8-15 = K(8h+16..8h+23), N = lane&15.
// ---------------------------------------------------------------------------
__global__ __launch_bounds__(192) void sa_out_gemm_kernel(
    const float* __restrict__ X, const unsigned short* __restrict__ wBF,
    float* __restrict__ out) {
  __align__(16) __shared__ unsigned short XT[64 * XPITCH];   // 37,888 B

  int t0  = blockIdx.x * 64;
  int b   = blockIdx.y;
  int tid = threadIdx.x;
  const float* Xb = X + (size_t)b * CDIM * TDIM;

  // Fill LDS: 288c x 64t, as 16 float4-quads per c-row; zeros for c>=272.
  for (int q = tid; q < CP * 16; q += 192) {
    int c  = q >> 4;
    int t4 = (q & 15) << 2;
    v4f v = {0.f, 0.f, 0.f, 0.f};
    if (c < CDIM) v = *(const v4f*)(Xb + (size_t)c * TDIM + t0 + t4);
    XT[(t4 + 0) * XPITCH + c] = f2bf(v.x);
    XT[(t4 + 1) * XPITCH + c] = f2bf(v.y);
    XT[(t4 + 2) * XPITCH + c] = f2bf(v.z);
    XT[(t4 + 3) * XPITCH + c] = f2bf(v.w);
  }
  __syncthreads();

  int wv = tid >> 5;                  // 0..5  -> d-tiles {3wv, 3wv+1, 3wv+2}
  int L = tid & 31, l = L & 15, h = L >> 4;

  v8f acc[3][4];
  #pragma unroll
  for (int i = 0; i < 3; ++i)
    #pragma unroll
    for (int n = 0; n < 4; ++n) acc[i][n] = (v8f){};

  #pragma unroll 1
  for (int kc = 0; kc < 9; ++kc) {
    int c0 = kc * 32;
    BF16x16 Af[3];
    #pragma unroll
    for (int i = 0; i < 3; ++i) {
      int d = (wv * 3 + i) * 16 + l;
      const unsigned short* wp = wBF + (size_t)d * CP + c0 + 8 * h;
      Af[i].q[0] = *(const v4u*)wp;          // K = 8h .. 8h+7
      Af[i].q[1] = *(const v4u*)(wp + 16);   // K = 8h+16 .. 8h+23
    }
    #pragma unroll
    for (int n = 0; n < 4; ++n) {
      BF16x16 Bf;
      const unsigned short* xp = &XT[(n * 16 + l) * XPITCH + c0 + 8 * h];
      Bf.q[0] = *(const v4u*)xp;             // K = 8h .. 8h+7
      Bf.q[1] = *(const v4u*)(xp + 16);      // K = 8h+16 .. 8h+23
      #pragma unroll
      for (int i = 0; i < 3; ++i)
        acc[i][n] = __builtin_amdgcn_wmma_f32_16x16x32_bf16(
            false, Af[i].v, false, Bf.v, (short)0, acc[i][n], false, false);
    }
  }

  // Store: VGPR r -> d = dtile*16 + r + 8h, t = t0 + n*16 + (lane&15)
  #pragma unroll
  for (int i = 0; i < 3; ++i) {
    int dbase = (wv * 3 + i) * 16 + 8 * h;
    #pragma unroll
    for (int n = 0; n < 4; ++n) {
      int t = t0 + n * 16 + l;
      #pragma unroll
      for (int r = 0; r < 8; ++r) {
        int d = dbase + r;
        if (d < DDIM) out[((size_t)b * DDIM + d) * TDIM + t] = acc[i][n][r];
      }
    }
  }
}

// ---------------------------------------------------------------------------
extern "C" void kernel_launch(void* const* d_in, const int* in_sizes, int n_in,
                              void* d_out, int out_size, void* d_ws, size_t ws_size,
                              hipStream_t stream) {
  const float* X   = (const float*)d_in[0];   // (128,272,1024)
  const float* loc = (const float*)d_in[1];   // (272,2)
  const float* zre = (const float*)d_in[2];   // (270,32,32)
  const float* zim = (const float*)d_in[3];   // (270,32,32)
  float* out = (float*)d_out;                 // (128,270,1024)

  char* ws = (char*)d_ws;
  float* cs = (float*)(ws);                                   // 1024*288*4 = 1,179,648
  float* sn = (float*)(ws + 1179648);                         // 1,179,648
  float* A  = (float*)(ws + 2359296);                         // 288*288*4 = 331,776
  unsigned short* wBF = (unsigned short*)(ws + 2691072);      // 288*288*2 = 165,888

  sa_phase_kernel<<<KL, CP, 0, stream>>>(loc, cs, sn);
  sa_a_wmma_kernel<<<18 * 18, 32, 0, stream>>>(zre, zim, cs, sn, A);
  sa_softmax_kernel<<<DP, 32, 0, stream>>>(A, wBF);
  sa_out_gemm_kernel<<<dim3(TDIM / 64, BSZ), 192, 0, stream>>>(X, wBF, out);
}
